// CovaBlock_635655160226
// MI455X (gfx1250) — matrix operand order repaired and checked by reading, not compiled, not run on hardware
//
#include <hip/hip_runtime.h>

typedef __bf16 bf16;
typedef __attribute__((ext_vector_type(16))) __bf16 v16bf;
typedef __attribute__((ext_vector_type(8)))  __bf16 bf16x8;
typedef __attribute__((ext_vector_type(4)))  __bf16 bf16x4;
typedef __attribute__((ext_vector_type(2)))  __bf16 bf16x2;
typedef __attribute__((ext_vector_type(8)))  float   v8f;
typedef __attribute__((ext_vector_type(4)))  float   f32x4;

#if defined(__has_builtin)
#if __has_builtin(__builtin_amdgcn_cvt_pk_bf16_f32)
#define HAVE_PK_BF16 1
#endif
#endif

// Let the backend lower fptrunc -> native bf16 convert if the ISA has one.
static __device__ __forceinline__ bf16 f2bf(float f) { return static_cast<bf16>(f); }
static __device__ __forceinline__ float bf2f(bf16 b) {
  unsigned short s = __builtin_bit_cast(unsigned short, b);
  unsigned u = ((unsigned)s) << 16;
  return __builtin_bit_cast(float, u);
}
static __device__ __forceinline__ bf16x2 f2bf_pk(float a, float b) {
#ifdef HAVE_PK_BF16
  return __builtin_amdgcn_cvt_pk_bf16_f32(a, b);
#else
  bf16x2 r; r[0] = f2bf(a); r[1] = f2bf(b); return r;
#endif
}
static __device__ __forceinline__ v16bf cat16(bf16x8 lo, bf16x8 hi) {
  return __builtin_shufflevector(lo, hi, 0, 1, 2, 3, 4, 5, 6, 7,
                                         8, 9, 10, 11, 12, 13, 14, 15);
}

// ---------------- per-(w,c) support means: 2048 rows of 8192 ----------------
__global__ __launch_bounds__(256) void mean_kernel(const float* __restrict__ x2,
                                                   float* __restrict__ mean) {
  int wid = threadIdx.x >> 5, lane = threadIdx.x & 31;
  int row = blockIdx.x * 8 + wid;             // 0..2047 ; w = row>>7, c = row&127
  int w = row >> 7, c = row & 127;
  const float* base = x2 + ((size_t)(w * 8) * 128 + c) * 1024;
  float s = 0.f;
  for (int sh = 0; sh < 8; ++sh) {
    const float* p = base + (size_t)sh * 128 * 1024;
    for (int i = lane; i < 1024; i += 32) s += p[i];
  }
  s += __shfl_xor(s, 16, 32); s += __shfl_xor(s, 8, 32); s += __shfl_xor(s, 4, 32);
  s += __shfl_xor(s, 2, 32);  s += __shfl_xor(s, 1, 32);
  if (lane == 0) mean[row] = s * (1.0f / 8192.0f);
}

// ---------------- per-(b,c) query inverse L2 norm: 32768 rows of 1024 --------
__global__ __launch_bounds__(256) void invn_kernel(const float* __restrict__ x1,
                                                   float* __restrict__ invn) {
  int wid = threadIdx.x >> 5, lane = threadIdx.x & 31;
  int row = blockIdx.x * 8 + wid;             // 0..32767
  const float* base = x1 + (size_t)row * 1024;
  float s = 0.f;
  for (int i = lane; i < 1024; i += 32) { float v = base[i]; s += v * v; }
  s += __shfl_xor(s, 16, 32); s += __shfl_xor(s, 8, 32); s += __shfl_xor(s, 4, 32);
  s += __shfl_xor(s, 2, 32);  s += __shfl_xor(s, 1, 32);
  if (lane == 0) invn[row] = rsqrtf(s);
}

// ------ per-class partial Gram (128x128, K split 4 ways) via bf16 WMMA ------
// grid: (16 classes, 4 k-parts); each part covers 2 shots (K = 2048).
// Each 128x32 f32 k-chunk is converted to bf16 ONCE into LDS (shared by all
// 8 waves), then A/B fragments are aligned ds_load_b128 reads.
#define TSTRIDE 40    // 32 + 8 bf16 pad: 80-B rows -> conflict-free 16-B reads
__global__ __launch_bounds__(256) void gram_kernel(const float* __restrict__ x2,
                                                   float* __restrict__ Gpart) {
  __shared__ bf16 tile[128 * TSTRIDE];        // [c][k] bf16
  int w = blockIdx.x, part = blockIdx.y;
  int tid = threadIdx.x;
  int wid = tid >> 5, lane = tid & 31;
  int half = lane >> 4, lan = lane & 15;
  const float* supw = x2 + (size_t)w * 8 * 128 * 1024;
  v8f acc[8];
  for (int nt = 0; nt < 8; ++nt)
    for (int r = 0; r < 8; ++r) acc[nt][r] = 0.f;

  for (int si = 0; si < 2; ++si) {
    const float* sb = supw + (size_t)(part * 2 + si) * 128 * 1024;
    for (int kb = 0; kb < 1024; kb += 32) {
      __syncthreads();                        // protect previous chunk's reads
      for (int it = 0; it < 4; ++it) {        // stage 128x32: 16 elems/thread
        int item = it * 256 + tid;            // 0..1023
        int c = item >> 3, k4 = (item & 7) * 4;
        f32x4 v = *(const f32x4*)(sb + (size_t)c * 1024 + kb + k4);
        bf16x4 t = __builtin_convertvector(v, bf16x4);
        *(bf16x4*)(tile + c * TSTRIDE + k4) = t;
      }
      __syncthreads();
      const bf16* ar = tile + (wid * 16 + lan) * TSTRIDE;   // A row: c
      bf16x8 alo = *(const bf16x8*)(ar + half * 16);
      bf16x8 ahi = *(const bf16x8*)(ar + half * 16 + 8);
      v16bf A = cat16(alo, ahi);
      for (int nt = 0; nt < 8; ++nt) {
        const bf16* br = tile + (nt * 16 + lan) * TSTRIDE + half * 16;
        bf16x8 b0 = *(const bf16x8*)(br);
        bf16x8 b1 = *(const bf16x8*)(br + 8);
        acc[nt] = __builtin_amdgcn_wmma_f32_16x16x32_bf16(
            false, A, false, cat16(b0, b1), (short)0, acc[nt], false, false);
      }
    }
  }
  float* Gw = Gpart + ((size_t)part * 16 + w) * 16384;
  for (int nt = 0; nt < 8; ++nt)
    for (int r = 0; r < 8; ++r)
      Gw[(size_t)(wid * 16 + half * 8 + r) * 128 + nt * 16 + lan] = acc[nt][r];
}

// ---- cova_bf16 = (sum_p Gpart - N m m^T) / (N-1), N = 8192 -----------------
__global__ __launch_bounds__(256) void cova_kernel(const float* __restrict__ Gpart,
                                                   const float* __restrict__ mean,
                                                   bf16* __restrict__ cova) {
  int idx = blockIdx.x * 256 + threadIdx.x;   // 262144
  int w = idx >> 14, c = (idx >> 7) & 127, d = idx & 127;
  float g = 0.f;
  for (int p = 0; p < 4; ++p) g += Gpart[(size_t)p * 16 * 16384 + idx];
  float v = (g - 8192.0f * mean[w * 128 + c] * mean[w * 128 + d]) * (1.0f / 8191.0f);
  cova[idx] = f2bf(v);
}

// ------------- main einsum: sim[b,w,i] = q_i^T Cova_w q_i --------------------
#define QSTRIDE 136   // 128 + 8 bf16 pad (272-B rows, 16-B aligned fragments)

static __device__ __forceinline__ void diag_reduce(
    const v8f* acc, const bf16* lds_q, float (*simbuf)[128], float* out,
    int w, int b, int i0, int tid, int wid, int half, int lan) {
  for (int jt = 0; jt < 8; ++jt) {
    int iloc = jt * 16 + lan;
    // q[d, i] for this wave's 8 M-rows: contiguous 8 bf16 -> one ds_load_b128
    bf16x8 qv = *(const bf16x8*)(lds_q + iloc * QSTRIDE + wid * 16 + half * 8);
    float ps = 0.f;
    for (int r = 0; r < 8; ++r) ps += acc[jt][r] * bf2f(qv[r]);
    ps += __shfl_xor(ps, 16, 32);            // merge the two M-halves
    if (half == 0) simbuf[wid][iloc] = ps;
  }
  __syncthreads();
  if (tid < 128) {
    float s = 0.f;
    for (int ww = 0; ww < 8; ++ww) s += simbuf[ww][tid];
    out[(size_t)b * 16384 + w * 1024 + i0 + tid] = s;
  }
  __syncthreads();
}

__global__ __launch_bounds__(256) void sim_kernel(const float* __restrict__ x1,
                                                  const bf16* __restrict__ cova,
                                                  const float* __restrict__ invn,
                                                  float* __restrict__ out) {
  __shared__ bf16 lds_q[128 * QSTRIDE];       // normalized query tile, [i][c]
  __shared__ float simbuf[8][128];
  int b  = blockIdx.y;
  int i0 = blockIdx.x * 128;
  int tid = threadIdx.x;
  int wid = tid >> 5, lane = tid & 31;
  int half = lane >> 4, lan = lane & 15;

  // stage q[b, :, i0..i0+127] * invnorm as bf16, transposed [i][c]
  for (int idx = tid; idx < 128 * 32; idx += 256) {
    int c = idx >> 5, i4 = (idx & 31) * 4;
    f32x4 v = *(const f32x4*)(x1 + ((size_t)b * 128 + c) * 1024 + i0 + i4);
    float sc = invn[b * 128 + c];
    f32x4 vs = v * sc;
    bf16x4 t = __builtin_convertvector(vs, bf16x4);
    lds_q[(i4 + 0) * QSTRIDE + c] = t[0];
    lds_q[(i4 + 1) * QSTRIDE + c] = t[1];
    lds_q[(i4 + 2) * QSTRIDE + c] = t[2];
    lds_q[(i4 + 3) * QSTRIDE + c] = t[3];
  }
  __syncthreads();

  // two classes per pass (wp, wp+8): each B fragment feeds 2 WMMAs
  for (int wp = 0; wp < 8; ++wp) {
    const bf16* rowA = cova + (size_t)wp * 16384 + (size_t)(wid * 16 + lan) * 128;
    const bf16* rowB = rowA + (size_t)8 * 16384;
    v8f acc0[8], acc1[8];
    for (int jt = 0; jt < 8; ++jt)
      for (int r = 0; r < 8; ++r) { acc0[jt][r] = 0.f; acc1[jt][r] = 0.f; }

    for (int kc = 0; kc < 4; ++kc) {
      int kb = kc * 32;
      bf16x8 a0lo = *(const bf16x8*)(rowA + kb + half * 16);
      bf16x8 a0hi = *(const bf16x8*)(rowA + kb + half * 16 + 8);
      bf16x8 a1lo = *(const bf16x8*)(rowB + kb + half * 16);
      bf16x8 a1hi = *(const bf16x8*)(rowB + kb + half * 16 + 8);
      v16bf A0 = cat16(a0lo, a0hi);           // A[m=d, k=c]
      v16bf A1 = cat16(a1lo, a1hi);
      for (int jt = 0; jt < 8; ++jt) {
        const bf16* bp = lds_q + (jt * 16 + lan) * QSTRIDE + kb + half * 16;
        bf16x8 b0 = *(const bf16x8*)(bp);
        bf16x8 b1 = *(const bf16x8*)(bp + 8);
        v16bf B = cat16(b0, b1);              // B[k=c, n=i]
        acc0[jt] = __builtin_amdgcn_wmma_f32_16x16x32_bf16(
            false, A0, false, B, (short)0, acc0[jt], false, false);
        acc1[jt] = __builtin_amdgcn_wmma_f32_16x16x32_bf16(
            false, A1, false, B, (short)0, acc1[jt], false, false);
      }
    }
    diag_reduce(acc0, lds_q, simbuf, out, wp,     b, i0, tid, wid, half, lan);
    diag_reduce(acc1, lds_q, simbuf, out, wp + 8, b, i0, tid, wid, half, lan);
  }
}

extern "C" void kernel_launch(void* const* d_in, const int* in_sizes, int n_in,
                              void* d_out, int out_size, void* d_ws, size_t ws_size,
                              hipStream_t stream) {
  (void)in_sizes; (void)n_in; (void)out_size; (void)ws_size;
  const float* x1 = (const float*)d_in[0];   // [256,128,32,32]
  const float* x2 = (const float*)d_in[1];   // [16,8,128,32,32]
  float* out = (float*)d_out;                // [256, 16*1024]

  char* ws = (char*)d_ws;
  float* mean  = (float*)ws;                                   //   8 KB
  float* invn  = (float*)(ws + 8192);                          // 128 KB
  float* Gpart = (float*)(ws + 8192 + 131072);                 //   4 MB (4 parts)
  bf16*  cova  = (bf16*) (ws + 8192 + 131072 + 4194304);       // 512 KB

  mean_kernel<<<256, 256, 0, stream>>>(x2, mean);
  invn_kernel<<<4096, 256, 0, stream>>>(x1, invn);
  gram_kernel<<<dim3(16, 4), 256, 0, stream>>>(x2, Gpart);
  cova_kernel<<<1024, 256, 0, stream>>>(Gpart, mean, cova);
  sim_kernel<<<dim3(8, 256), 256, 0, stream>>>(x1, cova, invn, out);
}
